// GlowGNN_72945724555617
// MI455X (gfx1250) — compile-verified, matrix operand order checked by reading, not run on hardware
//
#include <hip/hip_runtime.h>
#include <hip/hip_bf16.h>

// ---------------------------------------------------------------------------
// GlowGNN forward on MI455X (gfx1250): bf16 WMMA, weight-stationary edge MLP.
//   dims: N=50000 nodes, E=600000 edges, D=128, 8 layers, 44 etypes,
//         M=100000 mentions, V=20000 vars, OUT=100.
// ---------------------------------------------------------------------------

typedef __attribute__((ext_vector_type(16))) __bf16 v16bf;
typedef __attribute__((ext_vector_type(8)))  __bf16 v8bf;
typedef __attribute__((ext_vector_type(4)))  __bf16 v4bf;
typedef __attribute__((ext_vector_type(8)))  float  v8f;

#define DD 128
#define TWO_D 256
#define THREE_D 384
#define N_ETYPES 44
#define N_LAYERS 8
#define IN_DIM 64
#define OUT_DIM 100

// ---- WMMA fragment helpers (16x16x32 bf16) --------------------------------
// A-matrix: lane l<16 -> row M=l&15, kbase=0 ; l>=16 -> row l-16, kbase=8.
//   elems 0..7 = src[kb+0..7], elems 8..15 = src[kb+16..23] (32-wide K chunk)
// B-matrix (K x N) has the mirrored layout: lane -> col N, same K slicing.

__device__ inline v16bf frag_from_bf16(const __bf16* p) {
  v8bf lo = *(const v8bf*)(p);
  v8bf hi = *(const v8bf*)(p + 16);
  return __builtin_shufflevector(lo, hi, 0,1,2,3,4,5,6,7,8,9,10,11,12,13,14,15);
}

__device__ inline v16bf frag_from_f32(const float* p) {
  v16bf a;
#pragma unroll
  for (int j = 0; j < 8; ++j) { a[j] = (__bf16)p[j]; a[j + 8] = (__bf16)p[16 + j]; }
  return a;
}

__device__ inline v16bf frag_from_f32_scaled(const float* p, float s) {
  v16bf a;
#pragma unroll
  for (int j = 0; j < 8; ++j) { a[j] = (__bf16)(p[j] * s); a[j + 8] = (__bf16)(p[16 + j] * s); }
  return a;
}

__device__ inline v8f wmma_bf16(v16bf a, v16bf b, v8f c) {
  return __builtin_amdgcn_wmma_f32_16x16x32_bf16(false, a, false, b, (short)0, c, false, false);
}

// Packed fragments: buffer[((ntile*KC + c)*32 + lane)*16 + j], 32B per lane.
template <int KC>
__device__ inline v8f wmma_rowblock(const v16bf* a, const __bf16* wp, int lane) {
  v8f acc = {};
#pragma unroll
  for (int c = 0; c < KC; ++c) {
    v16bf b = *(const v16bf*)(wp + c * 512 + lane * 16);
    acc = wmma_bf16(a[c], b, acc);
  }
  return acc;
}

// ---- weight packing: f32 row-major [K][realN] -> bf16 fragment-major ------
__global__ void pack_frag_kernel(const float* __restrict__ W, __bf16* __restrict__ out,
                                 int K, int KC, int NT, int realN) {
  int idx = blockIdx.x * 256 + threadIdx.x;
  int total = NT * KC * 512;
  if (idx >= total) return;
  int j    = idx & 15;
  int lane = (idx >> 4) & 31;
  int rest = idx >> 9;
  int c    = rest % KC;
  int nt   = rest / KC;
  int n = nt * 16 + (lane & 15);
  int k = c * 32 + ((lane & 16) ? 8 : 0) + (j < 8 ? j : j + 8);
  float v = (n < realN && k < K) ? W[(size_t)k * realN + n] : 0.0f;
  out[idx] = (__bf16)v;
}

__global__ void f32_to_bf16_kernel(const float* __restrict__ in, __bf16* __restrict__ out, int n) {
  int i = blockIdx.x * 256 + threadIdx.x;
  if (i < n) out[i] = (__bf16)in[i];
}

__global__ void zero_f32_kernel(float* __restrict__ p, int n) {
  int i = blockIdx.x * 256 + threadIdx.x;
  if (i < n) p[i] = 0.0f;
}

// ---- encoder: h = relu(X @ W0 + b0) @ W1 + b1, write bf16 h ---------------
__global__ __launch_bounds__(128) void encoder_kernel(
    const float* __restrict__ X, const __bf16* __restrict__ w0p, const float* __restrict__ b0,
    const __bf16* __restrict__ w1p, const float* __restrict__ b1,
    __bf16* __restrict__ hb, int nNodes, int nTiles) {
  __shared__ __attribute__((aligned(32))) __bf16 tbuf[4][16 * DD];
  int lane = threadIdx.x & 31, wave = threadIdx.x >> 5;
  int tile = blockIdx.x * 4 + wave;
  int t = tile < nTiles ? tile : nTiles - 1;
  int row = lane & 15;
  int kb = (lane & 16) ? 8 : 0;
  int rb = kb;
  int colT = lane & 15;
  int node = t * 16 + row; if (node >= nNodes) node = nNodes - 1;
  const float* xp = X + (size_t)node * IN_DIM;
  v16bf a[2];
  a[0] = frag_from_f32(xp + kb);
  a[1] = frag_from_f32(xp + 32 + kb);
  for (int n = 0; n < 8; ++n) {
    v8f acc = wmma_rowblock<2>(a, w0p + (size_t)n * 2 * 512, lane);
    float bias = b0[n * 16 + colT];
#pragma unroll
    for (int r = 0; r < 8; ++r) {
      float v = acc[r] + bias; v = v > 0.0f ? v : 0.0f;
      tbuf[wave][(r + rb) * DD + n * 16 + colT] = (__bf16)v;
    }
  }
  __syncthreads();
  v16bf at[4];
#pragma unroll
  for (int c = 0; c < 4; ++c) at[c] = frag_from_bf16(&tbuf[wave][row * DD + c * 32 + kb]);
  for (int n = 0; n < 8; ++n) {
    v8f acc = wmma_rowblock<4>(at, w1p + (size_t)n * 4 * 512, lane);
    float bias = b1[n * 16 + colT];
#pragma unroll
    for (int r = 0; r < 8; ++r) {
      int nd = t * 16 + r + rb;
      if (tile < nTiles && nd < nNodes)
        hb[(size_t)nd * DD + n * 16 + colT] = (__bf16)(acc[r] + bias);
    }
  }
}

// ---- edge MLP, weight-stationary ------------------------------------------
// out1[n,e] = relu(sum_k W1[k,n] * m[e,k] + b1[n])       A = W1^T (resident)
// out2[n2,e] = sum_k' W2[k',n2] * t[e,k'] + b2[n2]        A = W2^T (resident)
// 512 threads = 16 waves; wave w owns GEMM1 n-tile w and GEMM2 n2-tile w&7.
// Block grid-strides over 32-edge pairs; m and t staged in LDS.
__global__ __launch_bounds__(512) void edge_mlp_kernel(
    const __bf16* __restrict__ hb, const __bf16* __restrict__ efb,
    const __bf16* __restrict__ w1p, const __bf16* __restrict__ w2p,
    const float* __restrict__ b1, const float* __restrict__ b2,
    const int* __restrict__ srcI, const int* __restrict__ dstI, const int* __restrict__ elab,
    float* __restrict__ acc, int nEdges, int nPairs) {
  __shared__ __attribute__((aligned(32))) __bf16 mS[32 * THREE_D];  // 24576 B
  __shared__ __attribute__((aligned(32))) __bf16 tS[32 * TWO_D];    // 16384 B
  __shared__ int dsts[32];
  const int tid  = threadIdx.x;
  const int lane = tid & 31;
  const int wave = tid >> 5;
  const int kb   = (lane & 16) ? 8 : 0;
  const int rb   = kb;
  const int colE = lane & 15;
  const int n2t  = wave & 7;
  const int et2  = wave >> 3;

  // ---- resident weight fragments + hoisted biases (once per block) ----
  v16bf aW1[12];
  {
    const __bf16* wb = w1p + (size_t)wave * 12 * 512;
#pragma unroll
    for (int c = 0; c < 12; ++c) aW1[c] = *(const v16bf*)(wb + c * 512 + lane * 16);
  }
  v16bf aW2[8];
  {
    const __bf16* wb = w2p + (size_t)n2t * 8 * 512;
#pragma unroll
    for (int c = 0; c < 8; ++c) aW2[c] = *(const v16bf*)(wb + c * 512 + lane * 16);
  }
  float bb1[8], bb2[8];
#pragma unroll
  for (int r = 0; r < 8; ++r) bb1[r] = b1[wave * 16 + rb + r];
#pragma unroll
  for (int r = 0; r < 8; ++r) bb2[r] = b2[n2t * 16 + rb + r];

  for (int pair = blockIdx.x; pair < nPairs; pair += gridDim.x) {
    const int base = pair * 32;

    // ---- stage m = [h[dst] | h[src] | ef] for 32 edges (16B chunks) ----
    // 32 edges * 3 segs * 128 bf16 = 1536 chunks of 16B; 3 per thread.
#pragma unroll
    for (int it = 0; it < 3; ++it) {
      int ci   = tid + it * 512;
      int erow = ci / 48;
      int pos  = ci - erow * 48;
      int seg  = pos >> 4;
      int o8   = (pos & 15) * 8;                 // bf16 elems within a 128-row
      int e  = base + erow;
      int ec = e < nEdges ? e : nEdges - 1;
      const __bf16* p;
      if (seg == 0)      p = hb  + (size_t)dstI[ec] * DD;
      else if (seg == 1) p = hb  + (size_t)srcI[ec] * DD;
      else               p = efb + (size_t)elab[ec] * DD;
      *(uint4*)&mS[erow * THREE_D + seg * DD + o8] = *(const uint4*)(p + o8);
    }
    if (tid < 32) {
      int e = base + tid;
      dsts[tid] = (e < nEdges) ? dstI[e] : -1;
      // look-ahead on the index streams (global_prefetch_b8)
      __builtin_prefetch(dstI + base + 32 * 8, 0, 1);
      __builtin_prefetch(srcI + base + 32 * 8, 0, 1);
    }
    __syncthreads();

    // ---- GEMM1: wave w -> n-tile w; two 16-edge halves ----
#pragma unroll
    for (int et = 0; et < 2; ++et) {
      const __bf16* mrow = &mS[(et * 16 + colE) * THREE_D];
      v8f cacc = {};
#pragma unroll
      for (int c = 0; c < 12; ++c)
        cacc = wmma_bf16(aW1[c], frag_from_bf16(mrow + c * 32 + kb), cacc);
      v8bf pk;
#pragma unroll
      for (int r = 0; r < 8; ++r) {
        float v = cacc[r] + bb1[r];
        pk[r] = (__bf16)(v > 0.0f ? v : 0.0f);
      }
      // C layout: n across VGPRs -> 8 contiguous bf16 per lane, one b128 store
      *(v8bf*)&tS[(et * 16 + colE) * TWO_D + wave * 16 + rb] = pk;
    }
    __syncthreads();

    // ---- GEMM2 + atomic segment-sum scatter ----
    {
      const int eloc = et2 * 16 + colE;
      const __bf16* trow = &tS[eloc * TWO_D];
      v8f cacc = {};
#pragma unroll
      for (int c = 0; c < 8; ++c)
        cacc = wmma_bf16(aW2[c], frag_from_bf16(trow + c * 32 + kb), cacc);
      int di = dsts[eloc];
      if (di >= 0) {
        float* ap = acc + (size_t)di * DD + n2t * 16 + rb;
#pragma unroll
        for (int r = 0; r < 8; ++r)
          atomicAdd(ap + r, cacc[r] + bb2[r]);
      }
    }
    __syncthreads();
  }
}

// ---- gelu (exact erf), vectorized x4: bf16 requantize + re-zero acc -------
__global__ void gelu_kernel(float* __restrict__ acc, __bf16* __restrict__ hb, int n4, int last) {
  int i = blockIdx.x * 256 + threadIdx.x;
  if (i >= n4) return;
  float4 x = *(const float4*)(acc + 4 * (size_t)i);
  float g0 = 0.5f * x.x * (1.0f + erff(x.x * 0.70710678118654752f));
  float g1 = 0.5f * x.y * (1.0f + erff(x.y * 0.70710678118654752f));
  float g2 = 0.5f * x.z * (1.0f + erff(x.z * 0.70710678118654752f));
  float g3 = 0.5f * x.w * (1.0f + erff(x.w * 0.70710678118654752f));
  v4bf h; h[0] = (__bf16)g0; h[1] = (__bf16)g1; h[2] = (__bf16)g2; h[3] = (__bf16)g3;
  *(v4bf*)(hb + 4 * (size_t)i) = h;
  float4 w;
  if (last) { w.x = g0; w.y = g1; w.z = g2; w.w = g3; }
  else      { w.x = 0.0f; w.y = 0.0f; w.z = 0.0f; w.w = 0.0f; }
  *(float4*)(acc + 4 * (size_t)i) = w;
}

// ---- readout: scatter-mean numerators + counts ----------------------------
__global__ void readout_kernel(const float* __restrict__ h, const int* __restrict__ gath,
                               const int* __restrict__ scat, float* __restrict__ sums,
                               float* __restrict__ cnts, int nM) {
  int idx = blockIdx.x * 256 + threadIdx.x;
  if (idx >= nM * DD) return;
  int m = idx >> 7, c = idx & (DD - 1);
  int node = gath[m], var = scat[m];
  atomicAdd(&sums[(size_t)var * DD + c], h[(size_t)node * DD + c]);
  if (c == 0) atomicAdd(&cnts[var], 1.0f);
}

// ---- decoder: y = relu(vr @ W0 + b0) @ Wl + bl ----------------------------
__global__ __launch_bounds__(128) void decoder_kernel(
    const float* __restrict__ sums, const float* __restrict__ cnts,
    const __bf16* __restrict__ w0p, const float* __restrict__ b0,
    const __bf16* __restrict__ wlp, const float* __restrict__ bl,
    float* __restrict__ y, int nVars, int nTiles) {
  __shared__ __attribute__((aligned(32))) __bf16 tbuf[4][16 * DD];
  int lane = threadIdx.x & 31, wave = threadIdx.x >> 5;
  int tile = blockIdx.x * 4 + wave;
  bool tileValid = tile < nTiles;
  int t = tileValid ? tile : 0;
  int row = lane & 15;
  int kb = (lane & 16) ? 8 : 0;
  int rb = kb;
  int colT = lane & 15;
  int var = t * 16 + row; if (var >= nVars) var = nVars - 1;
  float cnt = cnts[var]; cnt = cnt > 1.0f ? cnt : 1.0f;
  float inv = 1.0f / cnt;
  const float* sp = sums + (size_t)var * DD;
  v16bf a[4];
#pragma unroll
  for (int c = 0; c < 4; ++c) a[c] = frag_from_f32_scaled(sp + c * 32 + kb, inv);
  for (int n = 0; n < 8; ++n) {
    v8f av = wmma_rowblock<4>(a, w0p + (size_t)n * 4 * 512, lane);
    float bias = b0[n * 16 + colT];
#pragma unroll
    for (int r = 0; r < 8; ++r) {
      float v = av[r] + bias; v = v > 0.0f ? v : 0.0f;
      tbuf[wave][(r + rb) * DD + n * 16 + colT] = (__bf16)v;
    }
  }
  __syncthreads();
  v16bf at[4];
#pragma unroll
  for (int c = 0; c < 4; ++c) at[c] = frag_from_bf16(&tbuf[wave][row * DD + c * 32 + kb]);
  for (int n = 0; n < 7; ++n) {     // OUT padded to 112
    v8f av = wmma_rowblock<4>(at, wlp + (size_t)n * 4 * 512, lane);
    int col = n * 16 + colT;
    float bias = col < OUT_DIM ? bl[col] : 0.0f;
#pragma unroll
    for (int r = 0; r < 8; ++r) {
      int v2 = t * 16 + r + rb;
      if (tileValid && col < OUT_DIM && v2 < nVars)
        y[(size_t)v2 * OUT_DIM + col] = av[r] + bias;
    }
  }
}

// ---------------------------------------------------------------------------
extern "C" void kernel_launch(void* const* d_in, const int* in_sizes, int n_in,
                              void* d_out, int out_size, void* d_ws, size_t ws_size,
                              hipStream_t stream) {
  const float* node_labels = (const float*)d_in[0];
  const int*   edges       = (const int*)d_in[1];
  const int*   elab        = (const int*)d_in[2];
  const int*   vgath       = (const int*)d_in[3];
  const int*   vscat       = (const int*)d_in[4];
  const float* enc_W0 = (const float*)d_in[7];
  const float* enc_b0 = (const float*)d_in[8];
  const float* enc_W1 = (const float*)d_in[9];
  const float* enc_b1 = (const float*)d_in[10];
  const float* edge_emb = (const float*)d_in[11];
  const float* mlp_W1 = (const float*)d_in[12];
  const float* mlp_b1 = (const float*)d_in[13];
  const float* mlp_W2 = (const float*)d_in[14];
  const float* mlp_b2 = (const float*)d_in[15];
  const float* dec_W0 = (const float*)d_in[16];
  const float* dec_b0 = (const float*)d_in[17];
  const float* dec_Wl = (const float*)d_in[18];
  const float* dec_bl = (const float*)d_in[19];
  float* out = (float*)d_out;

  const int nNodes = in_sizes[0] / IN_DIM;       // 50000
  const int nEdges = in_sizes[1] / 2;            // 600000
  const int nMent  = in_sizes[3];                // 100000
  const int nVars  = out_size / OUT_DIM;         // 20000
  const int* src = edges;
  const int* dst = edges + nEdges;

  // ---- workspace carve-up ----
  char* ws = (char*)d_ws;
  size_t off = 0;
  auto carve = [&](size_t bytes) -> void* {
    void* p = ws + off;
    off = (off + bytes + 255) & ~(size_t)255;
    return p;
  };
  float*  accF  = (float*)carve((size_t)nNodes * DD * 4);
  __bf16* hBF   = (__bf16*)carve((size_t)nNodes * DD * 2);
  __bf16* efBF  = (__bf16*)carve((size_t)N_LAYERS * N_ETYPES * DD * 2);
  const size_t W1P_SZ = 16 * 12 * 512;           // 98304 frag elems per layer
  const size_t W2P_SZ = 8 * 8 * 512;             // 32768
  __bf16* w1P   = (__bf16*)carve(N_LAYERS * W1P_SZ * 2);
  __bf16* w2P   = (__bf16*)carve(N_LAYERS * W2P_SZ * 2);
  __bf16* encW0P = (__bf16*)carve((size_t)8 * 2 * 512 * 2);
  __bf16* encW1P = (__bf16*)carve((size_t)8 * 4 * 512 * 2);
  __bf16* decW0P = (__bf16*)carve((size_t)8 * 4 * 512 * 2);
  __bf16* decWlP = (__bf16*)carve((size_t)7 * 4 * 512 * 2);
  float*  varSum = (float*)carve((size_t)nVars * DD * 4);
  float*  varCnt = (float*)carve((size_t)nVars * 4);
  (void)ws_size;

  auto grid1d = [](int total) { return (total + 255) / 256; };

  // ---- pack weights to bf16 fragment layouts (every call; ws is volatile) ----
  pack_frag_kernel<<<grid1d(8 * 2 * 512), 256, 0, stream>>>(enc_W0, encW0P, IN_DIM, 2, 8, DD);
  pack_frag_kernel<<<grid1d(8 * 4 * 512), 256, 0, stream>>>(enc_W1, encW1P, DD, 4, 8, DD);
  for (int l = 0; l < N_LAYERS; ++l) {
    pack_frag_kernel<<<grid1d((int)W1P_SZ), 256, 0, stream>>>(
        mlp_W1 + (size_t)l * 3 * DD * 2 * DD, w1P + l * W1P_SZ, 3 * DD, 12, 16, 2 * DD);
    pack_frag_kernel<<<grid1d((int)W2P_SZ), 256, 0, stream>>>(
        mlp_W2 + (size_t)l * 2 * DD * DD, w2P + l * W2P_SZ, 2 * DD, 8, 8, DD);
  }
  pack_frag_kernel<<<grid1d(8 * 4 * 512), 256, 0, stream>>>(dec_W0, decW0P, DD, 4, 8, DD);
  pack_frag_kernel<<<grid1d(7 * 4 * 512), 256, 0, stream>>>(dec_Wl, decWlP, DD, 4, 7, OUT_DIM);
  f32_to_bf16_kernel<<<grid1d(N_LAYERS * N_ETYPES * DD), 256, 0, stream>>>(
      edge_emb, efBF, N_LAYERS * N_ETYPES * DD);

  // ---- zero accumulators ----
  zero_f32_kernel<<<grid1d(nNodes * DD), 256, 0, stream>>>(accF, nNodes * DD);
  zero_f32_kernel<<<grid1d(nVars * DD), 256, 0, stream>>>(varSum, nVars * DD);
  zero_f32_kernel<<<grid1d(nVars), 256, 0, stream>>>(varCnt, nVars);

  // ---- encoder ----
  {
    int tiles = (nNodes + 15) / 16;
    encoder_kernel<<<(tiles + 3) / 4, 128, 0, stream>>>(
        node_labels, encW0P, enc_b0, encW1P, enc_b1, hBF, nNodes, tiles);
  }

  // ---- 8 message-passing layers ----
  {
    int nPairs = (nEdges + 31) / 32;             // 18750 pairs of 16-edge tiles
    int blocks = nPairs < 1024 ? nPairs : 1024;
    int n4 = (nNodes * DD) / 4;
    for (int l = 0; l < N_LAYERS; ++l) {
      edge_mlp_kernel<<<blocks, 512, 0, stream>>>(
          hBF, efBF + (size_t)l * N_ETYPES * DD,
          w1P + l * W1P_SZ, w2P + l * W2P_SZ,
          mlp_b1 + (size_t)l * 2 * DD, mlp_b2 + (size_t)l * DD,
          src, dst, elab, accF, nEdges, nPairs);
      gelu_kernel<<<grid1d(n4), 256, 0, stream>>>(
          accF, hBF, n4, l == N_LAYERS - 1 ? 1 : 0);
    }
  }

  // ---- readout scatter-mean ----
  readout_kernel<<<grid1d(nMent * DD), 256, 0, stream>>>(accF, vgath, vscat, varSum, varCnt, nMent);

  // ---- decoder ----
  {
    int tiles = (nVars + 15) / 16;
    decoder_kernel<<<(tiles + 3) / 4, 128, 0, stream>>>(
        varSum, varCnt, decW0P, dec_b0, decWlP, dec_bl, out, nVars, tiles);
  }
}
